// Mamaba_68058051772536
// MI455X (gfx1250) — matrix operand order repaired and checked by reading, not compiled, not run on hardware
//
#include <hip/hip_runtime.h>
#include <hip/hip_bf16.h>
#include <stdint.h>

// ---- Model dims --------------------------------------------------------
#define BATCH   2
#define SEQ     1024
#define DM      768       // d_model (= 3*256)
#define DI      1536      // d_inner
#define NSTATE  16
#define KCONV   4
#define DTR     48
#define NLAYER  2
#define BBAT    4                 // direction-major batch: [fwd b0, fwd b1, bwd b0, bwd b1]
#define TTOK    (BBAT*SEQ)        // 4096 tokens through every GEMM
#define XPN     80
#define XPNPAD  128               // pad x_proj N to multiple of 64
#define LN_EPS  1e-5f

typedef _Float16 half_t;
typedef __attribute__((ext_vector_type(16))) _Float16 v16h;
typedef __attribute__((ext_vector_type(8)))  float    v8f;

union H2U { unsigned int u; _Float16 h[2]; };

__device__ __forceinline__ float sigmoidf_(float x){ return 1.f/(1.f+__expf(-x)); }
__device__ __forceinline__ float siluf_(float x){ return x*sigmoidf_(x); }
__device__ __forceinline__ float softplusf_(float x){ return (x>20.f)?x:log1pf(__expf(x)); }

// Block-wide mean/var over DM=768 (3 values per thread, 256 threads).
__device__ __forceinline__ void block_meanvar(float s, float sq, float* sh,
                                              float& mean, float& rstd){
  int tid = threadIdx.x;
  sh[tid] = s; sh[256+tid] = sq;
  __syncthreads();
  for (int off=128; off>0; off>>=1){
    if (tid < off){ sh[tid]+=sh[tid+off]; sh[256+tid]+=sh[256+tid+off]; }
    __syncthreads();
  }
  mean = sh[0]*(1.f/DM);
  float var = sh[256]*(1.f/DM) - mean*mean;
  rstd = rsqrtf(var + LN_EPS);
  __syncthreads();
}

// ---- WMMA GEMM: C[M,N] = A[M,K](f16,row) * W[N,K](f16,row)^T -----------
// Wave tile 32x64: 2 A frags + 4 B frags -> 8 WMMAs per K-step (B reused
// across two M-tiles). WG = 8 waves = 256x64 of C.
// Requires M%256==0, N%64==0, K%32==0 (all our GEMMs satisfy this).
__global__ void __launch_bounds__(256) gemm_wmma_f16(
    const half_t* __restrict__ A, const half_t* __restrict__ W,
    float* __restrict__ C, int M, int N, int K)
{
  (void)M;
  const int lane = threadIdx.x & 31;
  const int wave = threadIdx.x >> 5;
  const int hi   = lane >> 4;      // lane half (0/1)
  const int ln   = lane & 15;
  const int mrow0 = (blockIdx.y*16 + wave*2)*16;
  const int mrow1 = mrow0 + 16;
  const int ncol0 = blockIdx.x*64;

  const half_t* Arow0 = A + (size_t)(mrow0 + ln)*K;  // lane's M row, tile 0
  const half_t* Arow1 = A + (size_t)(mrow1 + ln)*K;  // lane's M row, tile 1
  const half_t* Wrow  = W + (size_t)(ncol0 + ln)*K;  // lane's N column (row of W)
  const size_t wstr = (size_t)16*K;

  v8f c00={},c01={},c02={},c03={},c10={},c11={},c12={},c13={};
  for (int k0 = 0; k0 < K; k0 += 32){
    __builtin_prefetch(Arow0 + k0 + 256, 0, 1);  // global_prefetch_b8
    __builtin_prefetch(Arow1 + k0 + 256, 0, 1);
    __builtin_prefetch(Wrow  + k0 + 256, 0, 1);
    // A fragments: 16x32, ISA layout (VGPR v: K = (v/4)*16 + half*8 + (v%4)*2)
    v16h a0, a1;
#pragma unroll
    for (int v = 0; v < 8; ++v){
      int kk = k0 + ((v>>2)<<4) + (hi<<3) + ((v&3)<<1);
      H2U t;
      t.u = *(const unsigned int*)(Arow0 + kk); a0[2*v]=t.h[0]; a0[2*v+1]=t.h[1];
      t.u = *(const unsigned int*)(Arow1 + kk); a1[2*v]=t.h[0]; a1[2*v+1]=t.h[1];
    }
    // B fragments: 32x16, ISA layout (VGPR v: K = (v/4)*16 + half*8 + (v%4)*2)
    v16h b0, b1, b2, b3;
#pragma unroll
    for (int v = 0; v < 8; ++v){
      int kk = k0 + ((v>>2)<<4) + (hi<<3) + ((v&3)<<1);
      H2U t;
      t.u = *(const unsigned int*)(Wrow + kk);          b0[2*v]=t.h[0]; b0[2*v+1]=t.h[1];
      t.u = *(const unsigned int*)(Wrow +   wstr + kk); b1[2*v]=t.h[0]; b1[2*v+1]=t.h[1];
      t.u = *(const unsigned int*)(Wrow + 2*wstr + kk); b2[2*v]=t.h[0]; b2[2*v+1]=t.h[1];
      t.u = *(const unsigned int*)(Wrow + 3*wstr + kk); b3[2*v]=t.h[0]; b3[2*v+1]=t.h[1];
    }
    c00 = __builtin_amdgcn_wmma_f32_16x16x32_f16(false,a0,false,b0,(short)0,c00,false,false);
    c10 = __builtin_amdgcn_wmma_f32_16x16x32_f16(false,a1,false,b0,(short)0,c10,false,false);
    c01 = __builtin_amdgcn_wmma_f32_16x16x32_f16(false,a0,false,b1,(short)0,c01,false,false);
    c11 = __builtin_amdgcn_wmma_f32_16x16x32_f16(false,a1,false,b1,(short)0,c11,false,false);
    c02 = __builtin_amdgcn_wmma_f32_16x16x32_f16(false,a0,false,b2,(short)0,c02,false,false);
    c12 = __builtin_amdgcn_wmma_f32_16x16x32_f16(false,a1,false,b2,(short)0,c12,false,false);
    c03 = __builtin_amdgcn_wmma_f32_16x16x32_f16(false,a0,false,b3,(short)0,c03,false,false);
    c13 = __builtin_amdgcn_wmma_f32_16x16x32_f16(false,a1,false,b3,(short)0,c13,false,false);
  }
  // D layout: VGPR v -> M = v + 8*half, N = lane%16
#pragma unroll
  for (int v = 0; v < 8; ++v){
    int r0 = mrow0 + v + (hi<<3);
    int r1 = mrow1 + v + (hi<<3);
    float* C0 = C + (size_t)r0*N + ncol0 + ln;
    float* C1 = C + (size_t)r1*N + ncol0 + ln;
    C0[0]=c00[v]; C0[16]=c01[v]; C0[32]=c02[v]; C0[48]=c03[v];
    C1[0]=c10[v]; C1[16]=c11[v]; C1[32]=c12[v]; C1[48]=c13[v];
  }
}

// ---- f32 -> f16 weight conversion with N padding -----------------------
__global__ void __launch_bounds__(256) cvt_pad_f16(
    const float* __restrict__ src, half_t* __restrict__ dst,
    int Ns, int Np, int K, int total)
{
  int i = blockIdx.x*256 + threadIdx.x;
  if (i >= total) return;
  int k = i % K; int r = i / K; int n = r % Np; int layer = r / Np;
  float v = (n < Ns) ? src[((size_t)layer*Ns + n)*K + k] : 0.f;
  dst[i] = (half_t)v;
}

// ---- Embedding gather --------------------------------------------------
__global__ void __launch_bounds__(256) embed_kernel(
    const int* __restrict__ ids, const float* __restrict__ em,
    float* __restrict__ h, int total)
{
  int i = blockIdx.x*256 + threadIdx.x;
  if (i >= total) return;
  int c = i % DM; int t = i / DM;
  h[i] = em[(size_t)ids[t]*DM + c];
}

// ---- Layer prep: resb[bb,l] = h+res (dir-mapped); xln16 = f16(LN(resb)) -
__global__ void __launch_bounds__(256) prep_kernel(
    const float* __restrict__ h, const float* __restrict__ res,
    const float* __restrict__ w, const float* __restrict__ b,
    float* __restrict__ resb, half_t* __restrict__ xln16)
{
  __shared__ float sh[512];
  int blk = blockIdx.x;             // 0..TTOK-1
  int bb = blk / SEQ, l = blk % SEQ;
  int bidx = bb & 1, dir = bb >> 1;
  int srcl = dir ? (SEQ-1-l) : l;
  size_t src = (size_t)(bidx*SEQ + srcl)*DM;
  size_t dst = (size_t)blk*DM;
  int tid = threadIdx.x;
  float v3[3]; float s=0.f, sq=0.f;
#pragma unroll
  for (int j=0;j<3;++j){ int c=tid+256*j; float v=h[src+c]+res[src+c];
    v3[j]=v; resb[dst+c]=v; s+=v; sq+=v*v; }
  float mean,rstd; block_meanvar(s,sq,sh,mean,rstd);
#pragma unroll
  for (int j=0;j<3;++j){ int c=tid+256*j;
    xln16[dst+c] = (half_t)((v3[j]-mean)*rstd*w[c]+b[c]); }
}

// ---- resb += add; o16 = f16(LN(resb)) ----------------------------------
__global__ void __launch_bounds__(256) addln_kernel(
    const float* __restrict__ add, float* __restrict__ resb,
    const float* __restrict__ w, const float* __restrict__ b,
    half_t* __restrict__ o16)
{
  __shared__ float sh[512];
  size_t base = (size_t)blockIdx.x*DM;
  int tid = threadIdx.x;
  float v3[3]; float s=0.f, sq=0.f;
#pragma unroll
  for (int j=0;j<3;++j){ int c=tid+256*j; float v=resb[base+c]+add[base+c];
    resb[base+c]=v; v3[j]=v; s+=v; sq+=v*v; }
  float mean,rstd; block_meanvar(s,sq,sh,mean,rstd);
#pragma unroll
  for (int j=0;j<3;++j){ int c=tid+256*j;
    o16[base+c] = (half_t)((v3[j]-mean)*rstd*w[c]+b[c]); }
}

// ---- Depthwise causal conv (K=4) + SiLU --------------------------------
__global__ void __launch_bounds__(256) conv_silu_kernel(
    const float* __restrict__ xz, const float* __restrict__ cw,
    const float* __restrict__ cb, float* __restrict__ xc,
    half_t* __restrict__ xc16)
{
  int i = blockIdx.x*256 + threadIdx.x;
  if (i >= TTOK*DI) return;
  int d = i % DI; int t = i / DI; int l = t % SEQ;
  float acc = cb[d];
#pragma unroll
  for (int k=0;k<KCONV;++k){
    int ll = l - (KCONV-1) + k;
    if (ll >= 0)
      acc += cw[d*KCONV + k] * xz[(size_t)(t-(KCONV-1)+k)*(2*DI) + d];
  }
  float v = siluf_(acc);
  xc[i] = v;
  xc16[i] = (half_t)v;
}

// ---- Fused selective scan ----------------------------------------------
// Grid: BBAT * (DI/16) blocks of 256. Wave = 2 channels * 16 states.
// Fuses dt-projection (K=48 via 16-lane xor reduce), softplus, recurrence,
// C-contraction, D skip, SiLU(z) gate; writes f16 for out_proj WMMA.
__global__ void __launch_bounds__(256) scan_kernel(
    const float* __restrict__ xdbl, const float* __restrict__ xc,
    const float* __restrict__ xz,   const float* __restrict__ dtw,
    const float* __restrict__ dtb,  const float* __restrict__ alog,
    const float* __restrict__ Dp,   half_t* __restrict__ yg16)
{
  const int lane = threadIdx.x & 31;
  const int wave = threadIdx.x >> 5;
  const int hi = lane >> 4;
  const int n  = lane & 15;
  const int nblk = DI/16;                           // 96
  const int bb = blockIdx.x / nblk;
  const int d  = (blockIdx.x % nblk)*16 + wave*2 + hi;

  const float An = -__expf(alog[(size_t)d*NSTATE + n]);
  const float w0 = dtw[(size_t)d*DTR + n];
  const float w1 = dtw[(size_t)d*DTR + n + 16];
  const float w2 = dtw[(size_t)d*DTR + n + 32];
  const float bias = dtb[d];
  const float Dd = Dp[d];

  float hst = 0.f;
  for (int l = 0; l < SEQ; ++l){
    const size_t t = (size_t)bb*SEQ + l;
    const float* xd = xdbl + t*XPNPAD;
    // delta = softplus(dt_row . x_dbl[:48] + bias), reduced over 16 state lanes
    float part = w0*xd[n] + w1*xd[n+16] + w2*xd[n+32];
    part += __shfl_xor(part,1); part += __shfl_xor(part,2);
    part += __shfl_xor(part,4); part += __shfl_xor(part,8);
    float delta = softplusf_(part + bias);
    float u = xc[t*DI + d];
    float dA = __expf(delta * An);
    hst = dA*hst + (delta * xd[48+n]) * u;          // B at cols 48..63
    float yp = hst * xd[64+n];                      // C at cols 64..79
    yp += __shfl_xor(yp,1); yp += __shfl_xor(yp,2);
    yp += __shfl_xor(yp,4); yp += __shfl_xor(yp,8);
    if (n == 0){
      float y = yp + u*Dd;
      float z = xz[t*(2*DI) + DI + d];
      yg16[t*DI + d] = (half_t)(y * siluf_(z));
    }
  }
}

// ---- GatedMLP gate: o16 = f16(y * silu(gate)) --------------------------
__global__ void __launch_bounds__(256) gate_kernel(
    const float* __restrict__ f, half_t* __restrict__ o)
{
  int i = blockIdx.x*256 + threadIdx.x;
  if (i >= TTOK*DI) return;
  int d = i % DI; size_t t = i / DI;
  float y = f[t*(2*DI) + d];
  float g = f[t*(2*DI) + DI + d];
  o[i] = (half_t)(y * siluf_(g));
}

// ---- Fold directions: h = LNf(out_f) + rev(LNf(out_b)); res = r_f+rev(r_b)
__global__ void __launch_bounds__(256) fold_kernel(
    const float* __restrict__ mix, const float* __restrict__ resb,
    const float* __restrict__ nw, const float* __restrict__ nb,
    float* __restrict__ h, float* __restrict__ res)
{
  __shared__ float sh[512];
  int blk = blockIdx.x;                 // 0..BATCH*SEQ-1
  int b = blk / SEQ, l = blk % SEQ;
  int tf = b*SEQ + l;
  int tb = (2+b)*SEQ + (SEQ-1-l);
  int tid = threadIdx.x;
  float vf[3], vb[3]; float s=0.f, sq=0.f;
#pragma unroll
  for (int j=0;j<3;++j){ int c=tid+256*j; float v=mix[(size_t)tf*DM+c];
    vf[j]=v; s+=v; sq+=v*v; }
  float mean,rstd; block_meanvar(s,sq,sh,mean,rstd);
#pragma unroll
  for (int j=0;j<3;++j) vf[j] = (vf[j]-mean)*rstd;
  s=0.f; sq=0.f;
#pragma unroll
  for (int j=0;j<3;++j){ int c=tid+256*j; float v=mix[(size_t)tb*DM+c];
    vb[j]=v; s+=v; sq+=v*v; }
  block_meanvar(s,sq,sh,mean,rstd);
#pragma unroll
  for (int j=0;j<3;++j){
    int c = tid+256*j;
    float hf = vf[j]*nw[c] + nb[c];
    float hb = (vb[j]-mean)*rstd*nw[c] + nb[c];
    size_t o = (size_t)(b*SEQ+l)*DM + c;
    h[o]   = hf + hb;
    res[o] = resb[(size_t)tf*DM+c] + resb[(size_t)tb*DM+c];
  }
}

// ---- Final: o = LN(h+res) per token ------------------------------------
__global__ void __launch_bounds__(256) finalln_kernel(
    const float* __restrict__ h, const float* __restrict__ r,
    const float* __restrict__ w, const float* __restrict__ b,
    float* __restrict__ o)
{
  __shared__ float sh[512];
  size_t base = (size_t)blockIdx.x*DM;
  int tid = threadIdx.x;
  float v3[3]; float s=0.f, sq=0.f;
#pragma unroll
  for (int j=0;j<3;++j){ int c=tid+256*j; float v=h[base+c]+r[base+c];
    v3[j]=v; s+=v; sq+=v*v; }
  float mean,rstd; block_meanvar(s,sq,sh,mean,rstd);
#pragma unroll
  for (int j=0;j<3;++j){ int c=tid+256*j; o[base+c]=(v3[j]-mean)*rstd*w[c]+b[c]; }
}

// ---- Mean over L (deterministic sequential sum per (b,c)) --------------
__global__ void __launch_bounds__(256) mean_kernel(
    const float* __restrict__ ln, float* __restrict__ out)
{
  int i = blockIdx.x*256 + threadIdx.x;
  if (i >= BATCH*DM) return;
  int b = i / DM, c = i % DM;
  float s = 0.f;
  for (int l = 0; l < SEQ; ++l) s += ln[((size_t)b*SEQ+l)*DM + c];
  out[i] = s * (1.f/SEQ);
}

// ========================================================================
extern "C" void kernel_launch(void* const* d_in, const int* in_sizes, int n_in,
                              void* d_out, int out_size, void* d_ws, size_t ws_size,
                              hipStream_t stream)
{
  (void)in_sizes; (void)n_in; (void)out_size; (void)ws_size;
  const int*   ids     = (const int*)  d_in[0];
  const float* embedw  = (const float*)d_in[1];
  const float* norm_w  = (const float*)d_in[2];
  const float* norm_b  = (const float*)d_in[3];
  const float* in_proj = (const float*)d_in[4];
  const float* conv_w  = (const float*)d_in[5];
  const float* conv_b  = (const float*)d_in[6];
  const float* x_proj  = (const float*)d_in[7];
  const float* dt_w    = (const float*)d_in[8];
  const float* dt_b    = (const float*)d_in[9];
  const float* A_log   = (const float*)d_in[10];
  const float* Dparam  = (const float*)d_in[11];
  const float* out_proj= (const float*)d_in[12];
  const float* norm2_w = (const float*)d_in[13];
  const float* norm2_b = (const float*)d_in[14];
  const float* fc1     = (const float*)d_in[15];
  const float* fc2     = (const float*)d_in[16];
  const float* normf_w = (const float*)d_in[17];
  const float* normf_b = (const float*)d_in[18];
  float* out = (float*)d_out;

  // Workspace carve (256B aligned)
  char* p = (char*)d_ws;
  auto take = [&](size_t bytes)->void*{
    void* r = (void*)p; p += (bytes + 255) & ~(size_t)255; return r; };
  half_t* w16_in  = (half_t*)take(sizeof(half_t)*(size_t)NLAYER*2*DI*DM);
  half_t* w16_xp  = (half_t*)take(sizeof(half_t)*(size_t)NLAYER*XPNPAD*DI);
  half_t* w16_out = (half_t*)take(sizeof(half_t)*(size_t)NLAYER*DM*DI);
  half_t* w16_fc1 = (half_t*)take(sizeof(half_t)*(size_t)NLAYER*2*DI*DM);
  half_t* w16_fc2 = (half_t*)take(sizeof(half_t)*(size_t)NLAYER*DM*DI);
  float*  hbuf  = (float*) take(sizeof(float)*(size_t)BATCH*SEQ*DM);
  float*  rbuf  = (float*) take(sizeof(float)*(size_t)BATCH*SEQ*DM);
  float*  resb  = (float*) take(sizeof(float)*(size_t)TTOK*DM);
  half_t* xln16 = (half_t*)take(sizeof(half_t)*(size_t)TTOK*DM);
  float*  xzb   = (float*) take(sizeof(float)*(size_t)TTOK*2*DI);
  float*  xcb   = (float*) take(sizeof(float)*(size_t)TTOK*DI);
  half_t* xc16  = (half_t*)take(sizeof(half_t)*(size_t)TTOK*DI);
  float*  xdbl  = (float*) take(sizeof(float)*(size_t)TTOK*XPNPAD);
  half_t* yg16  = (half_t*)take(sizeof(half_t)*(size_t)TTOK*DI);
  float*  mixb  = (float*) take(sizeof(float)*(size_t)TTOK*DM);
  half_t* x216  = (half_t*)take(sizeof(half_t)*(size_t)TTOK*DM);

  // Weights -> f16 (padded), every call (deterministic, no caching)
  auto cvt = [&](const float* src, half_t* dst, int Ns, int Np, int K){
    int tot = NLAYER*Np*K;
    cvt_pad_f16<<<(tot+255)/256, 256, 0, stream>>>(src, dst, Ns, Np, K, tot);
  };
  cvt(in_proj,  w16_in,  2*DI, 2*DI, DM);
  cvt(x_proj,   w16_xp,  XPN,  XPNPAD, DI);
  cvt(out_proj, w16_out, DM,   DM,   DI);
  cvt(fc1,      w16_fc1, 2*DI, 2*DI, DM);
  cvt(fc2,      w16_fc2, DM,   DM,   DI);

  {
    int tot = BATCH*SEQ*DM;
    embed_kernel<<<(tot+255)/256, 256, 0, stream>>>(ids, embedw, hbuf, tot);
  }
  hipMemsetAsync(rbuf, 0, sizeof(float)*(size_t)BATCH*SEQ*DM, stream);

  auto gemm = [&](const half_t* A, const half_t* W, float* C, int M, int N, int K){
    gemm_wmma_f16<<<dim3(N/64, M/256), 256, 0, stream>>>(A, W, C, M, N, K);
  };

  for (int layer = 0; layer < NLAYER; ++layer){
    prep_kernel<<<TTOK, 256, 0, stream>>>(hbuf, rbuf,
        norm_w + layer*DM, norm_b + layer*DM, resb, xln16);
    gemm(xln16, w16_in + (size_t)layer*2*DI*DM, xzb, TTOK, 2*DI, DM);
    conv_silu_kernel<<<(TTOK*DI+255)/256, 256, 0, stream>>>(xzb,
        conv_w + (size_t)layer*DI*KCONV, conv_b + (size_t)layer*DI, xcb, xc16);
    gemm(xc16, w16_xp + (size_t)layer*XPNPAD*DI, xdbl, TTOK, XPNPAD, DI);
    scan_kernel<<<BBAT*(DI/16), 256, 0, stream>>>(xdbl, xcb, xzb,
        dt_w  + (size_t)layer*DI*DTR,  dt_b + (size_t)layer*DI,
        A_log + (size_t)layer*DI*NSTATE, Dparam + (size_t)layer*DI, yg16);
    gemm(yg16, w16_out + (size_t)layer*DM*DI, mixb, TTOK, DM, DI);
    addln_kernel<<<TTOK, 256, 0, stream>>>(mixb, resb,
        norm2_w + layer*DM, norm2_b + layer*DM, x216);
    gemm(x216, w16_fc1 + (size_t)layer*2*DI*DM, xzb, TTOK, 2*DI, DM);
    gate_kernel<<<(TTOK*DI+255)/256, 256, 0, stream>>>(xzb, yg16);
    gemm(yg16, w16_fc2 + (size_t)layer*DM*DI, mixb, TTOK, DM, DI);
    fold_kernel<<<BATCH*SEQ, 256, 0, stream>>>(mixb, resb, normf_w, normf_b,
        hbuf, rbuf);
  }

  finalln_kernel<<<BATCH*SEQ, 256, 0, stream>>>(hbuf, rbuf, normf_w, normf_b, mixb);
  mean_kernel<<<(BATCH*DM+255)/256, 256, 0, stream>>>(mixb, out);
}
